// SpacetimeMHSA_31482110280225
// MI455X (gfx1250) — matrix operand order repaired and checked by reading, not compiled, not run on hardware
//
#include <hip/hip_runtime.h>
#include <hip/hip_bf16.h>
#include <math.h>

typedef __attribute__((ext_vector_type(16))) __bf16 v16bf;
typedef __attribute__((ext_vector_type(8)))  __bf16 v8bf;
typedef __attribute__((ext_vector_type(8)))  float  v8f;

#define HEADS    12
#define DIMHEAD  64
#define PATCHES  196
#define DIM      768
#define BATCH    8
#define FRAMES   16
#define TOKENS   (FRAMES * PATCHES + 1)   // 3137
#define MROWS    (BATCH * TOKENS)         // 25096
#define NQKV     (3 * HEADS * DIMHEAD)    // 2304
#define NKP      224                      // 197 keys padded to 14 wmma tiles

// ---------------------------------------------------------------------------
// WMMA fragment loaders (wave32, 16x16x32 bf16)
// A (MxK) layout: lane r=lane&15 is row; per lane 16 bf16 at K = {g*8..g*8+7,
// 16+g*8..16+g*8+7} with g=lane>>4.  Source LDS tile row-major [M][K].
// B (KxN) layout: lane r is column; per lane 16 bf16 at K = {g*8..g*8+7,
// 16+g*8..16+g*8+7} (same as A).
// Source LDS tile stored TRANSPOSED [N][K] so the 16 K values are contiguous.
// ---------------------------------------------------------------------------
__device__ __forceinline__ v16bf load_frag_mk(const __bf16* base, int ld,
                                              int row0, int k0, int r, int g) {
    const __bf16* p = base + (size_t)(row0 + r) * ld + k0 + g * 8;
    v8bf lo = *(const v8bf*)p;
    v8bf hi = *(const v8bf*)(p + 16);
    v16bf v;
#pragma unroll
    for (int i = 0; i < 8; ++i) { v[i] = lo[i]; v[i + 8] = hi[i]; }
    return v;
}

__device__ __forceinline__ v16bf load_frag_nk(const __bf16* base, int ld,
                                              int n0, int k0, int r, int g) {
    const __bf16* p = base + (size_t)(n0 + r) * ld + k0 + g * 8;
    v8bf lo = *(const v8bf*)p;
    v8bf hi = *(const v8bf*)(p + 16);
    v16bf v;
#pragma unroll
    for (int i = 0; i < 8; ++i) { v[i] = lo[i]; v[i + 8] = hi[i]; }
    return v;
}

// ---------------------------------------------------------------------------
// fp32 -> bf16 conversion (weights only; x converted inline in gemm)
// ---------------------------------------------------------------------------
__global__ void f2bf(const float* __restrict__ in, __bf16* __restrict__ out, long n) {
    long i = (long)blockIdx.x * blockDim.x + threadIdx.x;
    if (i < n) out[i] = (__bf16)in[i];
}

// ---------------------------------------------------------------------------
// Tiled GEMM: C[M,N] = A[M,K] * B[K,N].  256 threads = 8 waves.
// Block tile 64x128, k-step 32.  Wave w owns N-subtile w*16 and all four
// M-subtiles -> 4 WMMA per k-step sharing one B fragment.
// QKV==1: A is fp32 (x), outputs scattered as bf16 into q/k/v [b][h][t][d].
// QKV==0: A is bf16 (attn out), outputs stored fp32 row-major into C.
// ---------------------------------------------------------------------------
#define BM  64
#define LDA 40
#define LDB 40
template <int QKV>
__global__ __launch_bounds__(256) void gemm256(const void* __restrict__ Ap,
                                               const __bf16* __restrict__ Bp,
                                               float* __restrict__ C,
                                               __bf16* __restrict__ qb,
                                               __bf16* __restrict__ kb,
                                               __bf16* __restrict__ vb,
                                               int M, int N, int K) {
    __shared__ __bf16 As[BM * LDA];
    __shared__ __bf16 Bs[128 * LDB];

    const int m0  = blockIdx.y * BM;
    const int n0  = blockIdx.x * 128;
    const int tid = threadIdx.x;
    const int wave = tid >> 5, lane = tid & 31, g = lane >> 4, r = lane & 15;

    const int arow = tid >> 2;          // 0..63
    const int acol = (tid & 3) * 8;     // 0,8,16,24
    const int bk   = tid >> 3;          // 0..31
    const int bn   = (tid & 7) * 16;    // 0..112

    v8f acc[4];
#pragma unroll
    for (int s = 0; s < 4; ++s) acc[s] = (v8f){};

    for (int kk = 0; kk < K; kk += 32) {
        // stage A tile (rows m0..m0+63, k kk..kk+31); convert fp32->bf16 if needed
        if (QKV) {
            const float* A = (const float*)Ap;
            float4 va = make_float4(0.f, 0.f, 0.f, 0.f);
            float4 vbx = make_float4(0.f, 0.f, 0.f, 0.f);
            if (m0 + arow < M) {
                const float* pa = A + (size_t)(m0 + arow) * K + kk + acol;
                va  = *(const float4*)(pa);
                vbx = *(const float4*)(pa + 4);
            }
            __bf16* as = &As[arow * LDA + acol];
            as[0] = (__bf16)va.x;  as[1] = (__bf16)va.y;
            as[2] = (__bf16)va.z;  as[3] = (__bf16)va.w;
            as[4] = (__bf16)vbx.x; as[5] = (__bf16)vbx.y;
            as[6] = (__bf16)vbx.z; as[7] = (__bf16)vbx.w;
        } else {
            const __bf16* A = (const __bf16*)Ap;
            __bf16* as = &As[arow * LDA + acol];
            if (m0 + arow < M) {
                const __bf16* pa = A + (size_t)(m0 + arow) * K + kk + acol;
#pragma unroll
                for (int i = 0; i < 8; ++i) as[i] = pa[i];
            } else {
#pragma unroll
                for (int i = 0; i < 8; ++i) as[i] = (__bf16)0.0f;
            }
        }
        // stage B tile transposed into [n][k]
        const __bf16* pb = Bp + (size_t)(kk + bk) * N + n0 + bn;
#pragma unroll
        for (int i = 0; i < 16; ++i) Bs[(bn + i) * LDB + bk] = pb[i];
        __syncthreads();

        v16bf bm = load_frag_nk(Bs, LDB, wave * 16, 0, r, g);
#pragma unroll
        for (int s = 0; s < 4; ++s) {
            v16bf a = load_frag_mk(As, LDA, s * 16, 0, r, g);
            acc[s] = __builtin_amdgcn_wmma_f32_16x16x32_bf16(false, a, false, bm, (short)0, acc[s], false, false);
        }
        __syncthreads();
    }

    const int col = n0 + wave * 16 + r;
    if (QKV) {
        // col = d*36 + which*12 + h  (qkv.reshape(b,t,dh,3,h))
        const int d = col / 36, rem = col % 36, which = rem / 12, h = rem % 12;
        __bf16* dst = (which == 0) ? qb : (which == 1) ? kb : vb;
#pragma unroll
        for (int s = 0; s < 4; ++s) {
#pragma unroll
            for (int i = 0; i < 8; ++i) {
                int m = m0 + s * 16 + g * 8 + i;
                if (m < M) {
                    int b = m / TOKENS, t = m % TOKENS;
                    dst[((size_t)(b * HEADS + h) * TOKENS + t) * DIMHEAD + d] = (__bf16)acc[s][i];
                }
            }
        }
    } else {
#pragma unroll
        for (int s = 0; s < 4; ++s) {
#pragma unroll
            for (int i = 0; i < 8; ++i) {
                int m = m0 + s * 16 + g * 8 + i;
                if (m < M) C[(size_t)m * N + col] = acc[s][i];
            }
        }
    }
}

// ---------------------------------------------------------------------------
// Spatial attention: one block per (b, f, h).  197 keys (cls + 196 patches)
// padded to 224.  Whole K (row-major [j][d]) and V (transposed [d][j]) live in
// LDS; queries processed in 64-row chunks: S = QK^T (wmma) -> softmax (VALU)
// -> O = A V (wmma).  ~153 KB dynamic LDS => 2 workgroups / 320KB WGP.
// ---------------------------------------------------------------------------
#define ATTN_SMEM (64 * NKP * 4 + NKP * 72 * 2 + 64 * NKP * 2 + 64 * 72 * 2 + 64 * NKP * 2)

__global__ __launch_bounds__(256) void attn_spatial(const __bf16* __restrict__ qb,
                                                    const __bf16* __restrict__ kb,
                                                    const __bf16* __restrict__ vb,
                                                    __bf16* __restrict__ ao) {
    extern __shared__ char smem[];
    float*  Sc = (float*)smem;                                       // 64 x 224 f32
    __bf16* Kt = (__bf16*)(smem + 57344);                            // 224 x 72
    __bf16* Vt = (__bf16*)(smem + 57344 + 32256);                    // 64 x 224 (V^T)
    __bf16* Qc = (__bf16*)(smem + 57344 + 32256 + 28672);            // 64 x 72
    __bf16* Ac = (__bf16*)(smem + 57344 + 32256 + 28672 + 9216);     // 64 x 224

    const int id = blockIdx.x;
    const int b = id / (FRAMES * HEADS);
    const int rem = id % (FRAMES * HEADS);
    const int f = rem / HEADS;
    const int h = rem % HEADS;
    const size_t base = (size_t)(b * HEADS + h) * TOKENS * DIMHEAD;
    const __bf16* Kg = kb + base;
    const __bf16* Vg = vb + base;
    const __bf16* Qg = qb + base;

    const int tid = threadIdx.x;
    const int wave = tid >> 5, lane = tid & 31, g = lane >> 4, r = lane & 15;

    // load K row-major and V transposed; zero-pad rows 197..223
    if (tid < NKP) {
        const int j = tid;
        const int t = (j == 0) ? 0 : ((j <= PATCHES) ? (1 + f * PATCHES + (j - 1)) : -1);
#pragma unroll 8
        for (int d = 0; d < DIMHEAD; ++d) {
            __bf16 kv = (t >= 0) ? Kg[(size_t)t * DIMHEAD + d] : (__bf16)0.0f;
            __bf16 vv = (t >= 0) ? Vg[(size_t)t * DIMHEAD + d] : (__bf16)0.0f;
            Kt[j * 72 + d] = kv;
            Vt[d * NKP + j] = vv;
        }
    }
    __syncthreads();

    for (int c = 0; c < 4; ++c) {
        const int p0 = c * 64;
        {   // stage query chunk (zero-padded past 196)
            const int row = tid & 63, db = (tid >> 6) * 16;
            const int p = p0 + row;
            const int t = (p < PATCHES) ? (1 + f * PATCHES + p) : -1;
#pragma unroll
            for (int i = 0; i < 16; ++i)
                Qc[row * 72 + db + i] = (t >= 0) ? Qg[(size_t)t * DIMHEAD + db + i] : (__bf16)0.0f;
        }
        __syncthreads();

        // S = Q K^T  (4 x 14 tiles of 16x16, K-depth 64 = 2 wmma steps)
        for (int tt = wave; tt < 4 * 14; tt += 8) {
            const int mi = tt / 14, nj = tt % 14;
            v8f acc = {};
#pragma unroll
            for (int ks = 0; ks < 2; ++ks) {
                v16bf a  = load_frag_mk(Qc, 72, mi * 16, ks * 32, r, g);
                v16bf bm = load_frag_nk(Kt, 72, nj * 16, ks * 32, r, g);
                acc = __builtin_amdgcn_wmma_f32_16x16x32_bf16(false, a, false, bm, (short)0, acc, false, false);
            }
#pragma unroll
            for (int i = 0; i < 8; ++i)
                Sc[(mi * 16 + g * 8 + i) * NKP + nj * 16 + r] = acc[i] * 0.125f;
        }
        __syncthreads();

        // softmax over 197 valid columns, emit bf16 A (zeroed padding)
        if (tid < 64) {
            float* srow = Sc + tid * NKP;
            float mx = -1e30f;
            for (int j = 0; j < 197; ++j) mx = fmaxf(mx, srow[j]);
            float sum = 0.f;
            for (int j = 0; j < 197; ++j) { float e = __expf(srow[j] - mx); srow[j] = e; sum += e; }
            const float inv = 1.f / sum;
            const bool valid = (p0 + tid) < PATCHES;
            __bf16* arow = Ac + tid * NKP;
            for (int j = 0; j < NKP; ++j)
                arow[j] = (valid && j < 197) ? (__bf16)(srow[j] * inv) : (__bf16)0.0f;
        }
        __syncthreads();

        // O = A V  (4 x 4 tiles, K-depth 224 = 7 wmma steps)
        for (int tt = wave; tt < 16; tt += 8) {
            const int mi = tt / 4, nd = tt % 4;
            v8f acc = {};
#pragma unroll
            for (int ks = 0; ks < 7; ++ks) {
                v16bf a  = load_frag_mk(Ac, NKP, mi * 16, ks * 32, r, g);
                v16bf bm = load_frag_nk(Vt, NKP, nd * 16, ks * 32, r, g);
                acc = __builtin_amdgcn_wmma_f32_16x16x32_bf16(false, a, false, bm, (short)0, acc, false, false);
            }
            const int col = nd * 16 + r;
#pragma unroll
            for (int i = 0; i < 8; ++i) {
                const int p = p0 + mi * 16 + g * 8 + i;
                if (p < PATCHES) {
                    const int t = 1 + f * PATCHES + p;
                    ao[((size_t)b * TOKENS + t) * DIM + h * DIMHEAD + col] = (__bf16)acc[i];
                }
            }
        }
        __syncthreads();
    }
}

// ---------------------------------------------------------------------------
// CLS-token attention: one block per (b, h); query t=0 over all 3137 keys.
// ---------------------------------------------------------------------------
__global__ __launch_bounds__(256) void attn_cls(const __bf16* __restrict__ qb,
                                                const __bf16* __restrict__ kb,
                                                const __bf16* __restrict__ vb,
                                                __bf16* __restrict__ ao) {
    __shared__ float sbuf[TOKENS];
    __shared__ float qs[DIMHEAD];
    __shared__ float red[256];
    const int b = blockIdx.x / HEADS, h = blockIdx.x % HEADS;
    const size_t base = (size_t)(b * HEADS + h) * TOKENS * DIMHEAD;
    const int tid = threadIdx.x;

    if (tid < DIMHEAD) qs[tid] = (float)qb[base + tid];
    __syncthreads();

    float lmax = -1e30f;
    for (int j = tid; j < TOKENS; j += 256) {
        float s = 0.f;
#pragma unroll 8
        for (int d = 0; d < DIMHEAD; ++d) s += qs[d] * (float)kb[base + (size_t)j * DIMHEAD + d];
        s *= 0.125f;
        sbuf[j] = s;
        lmax = fmaxf(lmax, s);
    }
    red[tid] = lmax;
    __syncthreads();
    for (int off = 128; off > 0; off >>= 1) {
        if (tid < off) red[tid] = fmaxf(red[tid], red[tid + off]);
        __syncthreads();
    }
    const float mx = red[0];
    __syncthreads();

    float lsum = 0.f;
    for (int j = tid; j < TOKENS; j += 256) {
        float e = __expf(sbuf[j] - mx);
        sbuf[j] = e;
        lsum += e;
    }
    red[tid] = lsum;
    __syncthreads();
    for (int off = 128; off > 0; off >>= 1) {
        if (tid < off) red[tid] += red[tid + off];
        __syncthreads();
    }
    const float inv = 1.f / red[0];

    if (tid < DIMHEAD) {
        float acc = 0.f;
        for (int j = 0; j < TOKENS; ++j) acc += sbuf[j] * (float)vb[base + (size_t)j * DIMHEAD + tid];
        ao[(size_t)b * TOKENS * DIM + h * DIMHEAD + tid] = (__bf16)(acc * inv);
    }
}

// ---------------------------------------------------------------------------
// launch
// ---------------------------------------------------------------------------
extern "C" void kernel_launch(void* const* d_in, const int* in_sizes, int n_in,
                              void* d_out, int out_size, void* d_ws, size_t ws_size,
                              hipStream_t stream) {
    (void)in_sizes; (void)n_in; (void)out_size; (void)ws_size;
    const float* x    = (const float*)d_in[0];
    const float* Wqkv = (const float*)d_in[1];
    const float* W0   = (const float*)d_in[2];
    float* out = (float*)d_out;

    char* p = (char*)d_ws;
    auto carve = [&](size_t bytes) {
        char* q = p;
        p += (bytes + 255) & ~(size_t)255;
        return q;
    };
    __bf16* wqkv_b = (__bf16*)carve((size_t)DIM * NQKV * 2);
    __bf16* w0_b   = (__bf16*)carve((size_t)DIM * DIM * 2);
    __bf16* qbuf   = (__bf16*)carve((size_t)BATCH * HEADS * TOKENS * DIMHEAD * 2);
    __bf16* kbuf   = (__bf16*)carve((size_t)BATCH * HEADS * TOKENS * DIMHEAD * 2);
    __bf16* vbuf   = (__bf16*)carve((size_t)BATCH * HEADS * TOKENS * DIMHEAD * 2);
    __bf16* ao     = (__bf16*)carve((size_t)MROWS * DIM * 2);

    f2bf<<<((long)DIM * NQKV + 255) / 256, 256, 0, stream>>>(Wqkv, wqkv_b, (long)DIM * NQKV);
    f2bf<<<((long)DIM * DIM + 255) / 256, 256, 0, stream>>>(W0, w0_b, (long)DIM * DIM);

    dim3 g1(NQKV / 128, (MROWS + BM - 1) / BM);
    gemm256<1><<<g1, 256, 0, stream>>>(x, wqkv_b, nullptr, qbuf, kbuf, vbuf, MROWS, NQKV, DIM);

    hipFuncSetAttribute(reinterpret_cast<const void*>(attn_spatial),
                        hipFuncAttributeMaxDynamicSharedMemorySize, ATTN_SMEM);
    attn_spatial<<<BATCH * FRAMES * HEADS, 256, ATTN_SMEM, stream>>>(qbuf, kbuf, vbuf, ao);
    attn_cls<<<BATCH * HEADS, 256, 0, stream>>>(qbuf, kbuf, vbuf, ao);

    dim3 g2(DIM / 128, (MROWS + BM - 1) / BM);
    gemm256<0><<<g2, 256, 0, stream>>>(ao, w0_b, out, nullptr, nullptr, nullptr, MROWS, DIM, DIM);
}
